// GCN_21715354649832
// MI455X (gfx1250) — compile-verified, matrix operand order checked by reading, not run on hardware
//
#include <hip/hip_runtime.h>
#include <hip/hip_bf16.h>
#include <math.h>

typedef __attribute__((ext_vector_type(16))) _Float16 v16h;
typedef __attribute__((ext_vector_type(8)))  float    v8f;

// ---------------------------------------------------------------------------
// Utility kernels
// ---------------------------------------------------------------------------
__global__ void fill_kernel(float* __restrict__ p, float v, int n) {
    int i = blockIdx.x * blockDim.x + threadIdx.x;
    if (i < n) p[i] = v;
}

__global__ void deg_count_kernel(const int* __restrict__ col,
                                 float* __restrict__ deg, int E) {
    int e = blockIdx.x * blockDim.x + threadIdx.x;
    if (e < E) atomicAdd(&deg[col[e]], 1.0f);
}

__global__ void to_dinv_kernel(float* __restrict__ deg, int n) {
    int i = blockIdx.x * blockDim.x + threadIdx.x;
    if (i < n) deg[i] = __frsqrt_rn(deg[i]);   // deg >= 1 always (self loop)
}

// ---------------------------------------------------------------------------
// Pack W[K,Nout] (f32 row-major) into f16 WMMA B-fragment order:
//   Wp[((kt*NT + nt)*32 + lane)*16 + j] = (f16) W[k(j,lane) * Nout + n(lane)]
// with k(j,lane) = kt*32 + (j/8)*16 + (lane/16)*8 + (j%8), n = nt*16 + lane%16
// (CDNA5 ISA 7.12.2 16-bit fragment layout). One thread per f16 element.
// ---------------------------------------------------------------------------
__global__ void pack_w_kernel(const float* __restrict__ W,
                              _Float16* __restrict__ Wp,
                              int Nout, int NT, int total) {
    int idx = blockIdx.x * blockDim.x + threadIdx.x;
    if (idx >= total) return;
    const int j    = idx & 15;
    const int lane = (idx >> 4) & 31;
    const int tile = idx >> 9;
    const int nt   = tile % NT;
    const int kt   = tile / NT;
    const int k = kt * 32 + ((j >> 3) << 4) + ((lane >> 4) << 3) + (j & 7);
    const int n = nt * 16 + (lane & 15);
    Wp[idx] = (_Float16)W[(size_t)k * Nout + n];
}

// ---------------------------------------------------------------------------
// WMMA GEMM: H[M, NT*16] = A[M, K] @ W, f32 A loaded as 4x b128 + cvt,
// B pre-packed f16 fragments (one 32B v16h load per lane per K-step).
// One wave per 16x16 tile; blockDim.x = NT*32. M must be a multiple of 16.
// K, NT are compile-time -> fully unrolled WMMA chain.
// ---------------------------------------------------------------------------
template <int K, int NT>
__global__ void gemm_wmma_f16(const float* __restrict__ A,
                              const _Float16* __restrict__ Wp,
                              float* __restrict__ H, int M) {
    constexpr int Nout = NT * 16;
    const int wave = threadIdx.x >> 5;      // N-tile index within block
    const int lane = threadIdx.x & 31;
    const int half = lane >> 4;             // 0 | 1
    const int r    = lane & 15;
    const int m0   = blockIdx.x * 16;

    const float* arow = A + (size_t)(m0 + r) * K;
    const v16h*  bp   = (const v16h*)Wp;

    v8f acc = {};
#pragma unroll
    for (int kt = 0; kt < K / 32; ++kt) {
        // A fragment: K offsets {kt*32 + half*8 + 0..7, kt*32 + 16 + half*8 + 0..7}
        const float4* ap0 = (const float4*)(arow + kt * 32 + (half << 3));
        const float4* ap1 = (const float4*)(arow + kt * 32 + 16 + (half << 3));
        const float4 a0 = ap0[0], a1 = ap0[1];
        const float4 a2 = ap1[0], a3 = ap1[1];
        v16h a;
        a[0]  = (_Float16)a0.x; a[1]  = (_Float16)a0.y;
        a[2]  = (_Float16)a0.z; a[3]  = (_Float16)a0.w;
        a[4]  = (_Float16)a1.x; a[5]  = (_Float16)a1.y;
        a[6]  = (_Float16)a1.z; a[7]  = (_Float16)a1.w;
        a[8]  = (_Float16)a2.x; a[9]  = (_Float16)a2.y;
        a[10] = (_Float16)a2.z; a[11] = (_Float16)a2.w;
        a[12] = (_Float16)a3.x; a[13] = (_Float16)a3.y;
        a[14] = (_Float16)a3.z; a[15] = (_Float16)a3.w;

        const v16h b = bp[(size_t)(kt * NT + wave) * 32 + lane];

        acc = __builtin_amdgcn_wmma_f32_16x16x32_f16(
                  /*neg_a=*/false, a, /*neg_b=*/false, b,
                  /*c_mod=*/(short)0, acc, /*reuse_a=*/false, /*reuse_b=*/false);
    }

    // 32-bit C/D layout: VGPR i -> row M = m0 + i + 8*half, col = n0 + r
    float* hbase = H + (size_t)m0 * Nout + wave * 16 + r;
#pragma unroll
    for (int i = 0; i < 8; ++i)
        hbase[(size_t)(i + half * 8) * Nout] = acc[i];
}

// ---------------------------------------------------------------------------
// Edge scatter: one wave per edge; agg[col] += h[row] * dinv[row]*dinv[col]
// VEC = D/32 consecutive floats per lane (4 -> b128 gather, 2 -> b64).
// ---------------------------------------------------------------------------
template <int VEC>
__global__ void scatter_edges_kernel(const float* __restrict__ H,
                                     const int* __restrict__ row,
                                     const int* __restrict__ col,
                                     const float* __restrict__ dinv,
                                     float* __restrict__ AGG, int E) {
    constexpr int D = VEC * 32;
    const int e = blockIdx.x * (blockDim.x >> 5) + (threadIdx.x >> 5);
    if (e >= E) return;
    const int lane = threadIdx.x & 31;
    const int s = row[e];
    const int t = col[e];
    const float w = dinv[s] * dinv[t];
    const float* src = H   + (size_t)s * D + lane * VEC;
    float*       dst = AGG + (size_t)t * D + lane * VEC;
    __builtin_prefetch(src, 0, 3);          // global_prefetch_b8 (L2-resident)
    float v[VEC];
    if (VEC == 4) {
        const float4 q = *(const float4*)src;
        v[0] = q.x; v[1] = q.y; v[2] = q.z; v[3] = q.w;
    } else {
        const float2 q = *(const float2*)src;
        v[0] = q.x; v[1] = q.y;
    }
#pragma unroll
    for (int i = 0; i < VEC; ++i)
        atomicAdd(&dst[i], v[i] * w);
}

// ---------------------------------------------------------------------------
// Finalize: AGG = (AGG + H*dinv^2 + bias), optional ReLU (in place).
// Folds the self-loop contribution (norm = dinv[i]^2) and the bias.
// ---------------------------------------------------------------------------
__global__ void finalize_kernel(float* __restrict__ AGG,
                                const float* __restrict__ H,
                                const float* __restrict__ dinv,
                                const float* __restrict__ bias,
                                int N, int D, int do_relu) {
    const int idx = blockIdx.x * blockDim.x + threadIdx.x;
    if (idx >= N * D) return;
    const int i = idx / D;
    const int d = idx - i * D;
    const float di = dinv[i];
    float v = AGG[idx] + H[idx] * di * di + bias[d];
    if (do_relu) v = fmaxf(v, 0.0f);
    AGG[idx] = v;
}

// ---------------------------------------------------------------------------
// Head: out[i] = sigmoid(dot(R[i, 0:64], Wfc) + bfc); one wave per node.
// ---------------------------------------------------------------------------
__global__ void head_kernel(const float* __restrict__ R,
                            const float* __restrict__ Wfc,
                            const float* __restrict__ bfc,
                            float* __restrict__ out, int N, int D) {
    const int i = blockIdx.x * (blockDim.x >> 5) + (threadIdx.x >> 5);
    if (i >= N) return;
    const int lane = threadIdx.x & 31;
    float s = 0.0f;
    for (int d = lane; d < D; d += 32)
        s += R[(size_t)i * D + d] * Wfc[d];
#pragma unroll
    for (int off = 16; off > 0; off >>= 1)
        s += __shfl_down(s, off, 32);
    if (lane == 0)
        out[i] = 1.0f / (1.0f + __expf(-(s + bfc[0])));
}

// ---------------------------------------------------------------------------
// Launch
// ---------------------------------------------------------------------------
extern "C" void kernel_launch(void* const* d_in, const int* in_sizes, int n_in,
                              void* d_out, int out_size, void* d_ws, size_t ws_size,
                              hipStream_t stream) {
    const float* x   = (const float*)d_in[0];
    const int*   ei  = (const int*)  d_in[1];
    const float* W1  = (const float*)d_in[2];
    const float* b1  = (const float*)d_in[3];
    const float* W2  = (const float*)d_in[4];
    const float* b2  = (const float*)d_in[5];
    const float* Wfc = (const float*)d_in[6];
    const float* bfc = (const float*)d_in[7];
    float* out = (float*)d_out;

    const int DIN = 256;
    const int N   = in_sizes[0] / DIN;   // 50000 (multiple of 16)
    const int E   = in_sizes[1] / 2;     // 800000
    const int DH  = 128;
    const int DE  = 64;

    const int* row = ei;                 // edge_index[0] (sources)
    const int* col = ei + E;             // edge_index[1] (targets)

    // Workspace layout (256B-aligned slices)
    char* ws = (char*)d_ws;
    size_t off = 0;
    auto take = [&](size_t bytes) -> void* {
        void* p = (void*)(ws + off);
        off += (bytes + 255) & ~(size_t)255;
        return p;
    };
    float*    dinv = (float*)   take((size_t)N * sizeof(float));
    float*    h1   = (float*)   take((size_t)N * DH * sizeof(float));
    float*    agg1 = (float*)   take((size_t)N * DH * sizeof(float));
    float*    h2   = (float*)   take((size_t)N * DE * sizeof(float));
    float*    agg2 = (float*)   take((size_t)N * DE * sizeof(float));
    _Float16* Wp1  = (_Float16*)take((size_t)DIN * DH * sizeof(_Float16));
    _Float16* Wp2  = (_Float16*)take((size_t)DH  * DE * sizeof(_Float16));

    const int T = 256;

    // Normalization: deg = 1 (self loop) + scatter count; dinv = rsqrt(deg)
    fill_kernel     <<<(N + T - 1) / T, T, 0, stream>>>(dinv, 1.0f, N);
    deg_count_kernel<<<(E + T - 1) / T, T, 0, stream>>>(col, dinv, E);
    to_dinv_kernel  <<<(N + T - 1) / T, T, 0, stream>>>(dinv, N);

    // Pack weights into f16 WMMA B-fragment order (tiny, once per launch)
    pack_w_kernel<<<(DIN * DH + T - 1) / T, T, 0, stream>>>(W1, Wp1, DH, DH / 16, DIN * DH);
    pack_w_kernel<<<(DH  * DE + T - 1) / T, T, 0, stream>>>(W2, Wp2, DE, DE / 16, DH * DE);

    // Layer 1: h1 = x @ W1 ; agg1 = scatter(norm * h1) + self + b1 ; relu
    gemm_wmma_f16<256, 8><<<N / 16, 8 * 32, 0, stream>>>(x, Wp1, h1, N);
    fill_kernel<<<((N * DH) + T - 1) / T, T, 0, stream>>>(agg1, 0.0f, N * DH);
    scatter_edges_kernel<4><<<(E + 7) / 8, T, 0, stream>>>(h1, row, col, dinv, agg1, E);
    finalize_kernel<<<((N * DH) + T - 1) / T, T, 0, stream>>>(agg1, h1, dinv, b1, N, DH, 1);

    // Layer 2: h2 = relu1 @ W2 ; agg2 = scatter + self + b2
    gemm_wmma_f16<128, 4><<<N / 16, 4 * 32, 0, stream>>>(agg1, Wp2, h2, N);
    fill_kernel<<<((N * DE) + T - 1) / T, T, 0, stream>>>(agg2, 0.0f, N * DE);
    scatter_edges_kernel<2><<<(E + 7) / 8, T, 0, stream>>>(h2, row, col, dinv, agg2, E);
    finalize_kernel<<<((N * DE) + T - 1) / T, T, 0, stream>>>(agg2, h2, dinv, b2, N, DE, 0);

    // Head: sigmoid(agg2 @ Wfc + bfc)
    head_kernel<<<(N + 7) / 8, T, 0, stream>>>(agg2, Wfc, bfc, out, N, DE);
}
